// Noise_68307159875957
// MI455X (gfx1250) — compile-verified, matrix operand order checked by reading, not run on hardware
//
#include <hip/hip_runtime.h>
#include <hip/hip_bf16.h>

// ---------------------------------------------------------------------------
// Problem constants (from reference)
// ---------------------------------------------------------------------------
#define Bc   16
#define Tc   2048
#define Cc   512
#define Hc   256
#define NOc  5
#define Rr   8          // gaussian radius, kernel = 17 taps

typedef __attribute__((ext_vector_type(16))) __bf16 v16bf;
typedef __attribute__((ext_vector_type(8)))  __bf16 v8bf;
typedef __attribute__((ext_vector_type(8)))  float  v8f;

// f32 -> bf16 bits, round-to-nearest-even
static __device__ __forceinline__ unsigned bf16bits(float f) {
    unsigned u = __float_as_uint(f);
    return (u + 0x7FFFu + ((u >> 16) & 1u)) >> 16;
}
// pack two f32 -> one dword of 2 bf16 (lo in low 16 bits)
static __device__ __forceinline__ unsigned pack_bf16(float lo, float hi) {
    return (bf16bits(lo) & 0xFFFFu) | (bf16bits(hi) << 16);
}

static __device__ __forceinline__ float gelu_exact(float x) {
    return 0.5f * x * (1.0f + erff(x * 0.70710678118654752f));
}

// ---------------------------------------------------------------------------
// Phase 0: zero the smooths accumulator
// ---------------------------------------------------------------------------
__global__ void noise_init_kernel(float* smacc) {
    if (threadIdx.x < 128) smacc[threadIdx.x] = 0.0f;
}

// ---------------------------------------------------------------------------
// Phase 1: fused bf16-WMMA GEMM
//   h = gelu(x @ w1[head] + b1[head])           (never materialized)
//   muSig[b,t,head,0:2] = h @ w2 + b2           (per-block complete: full H)
//   smacc[b,head]      += sum_t  h . ws         (global f32 atomics)
//
// block = 256 threads (8 waves). blockIdx.x = mtile*NO + head.
// Each block: 16 M-rows (contiguous t within one b) x all 256 H-cols.
// Each wave: two 16x16 N-tiles, K-loop of 16 steps of v_wmma_f32_16x16x32_bf16.
// ---------------------------------------------------------------------------
__global__ void __launch_bounds__(256)
noise_gemm_wmma_kernel(const float* __restrict__ x,
                       const float* __restrict__ w1,
                       const float* __restrict__ b1,
                       const float* __restrict__ w2,
                       const float* __restrict__ b2,
                       const float* __restrict__ wsv,
                       float* __restrict__ muSig,
                       float* __restrict__ smacc)
{
    // LDS staging (bf16): A tile 16x512 (row stride 520), B tile 256x32 (stride 40)
    __shared__ __bf16 As[16 * 520];     // 16.6 KB
    __shared__ __bf16 Bs[256 * 40];     // 20.5 KB
    __shared__ float  muRow[32];        // 16 rows x {mu,sig}
    __shared__ float  smLds;

    const int tid  = threadIdx.x;
    const int lane = tid & 31;
    const int wv   = tid >> 5;          // 0..7
    const int blk  = blockIdx.x;
    const int head = blk % NOc;
    const int mtile= blk / NOc;

    const int r0 = mtile * 16;          // global row (b*T + t)
    const int b  = r0 >> 11;            // / Tc
    const int t0 = r0 & (Tc - 1);

    if (tid < 32) muRow[tid] = 0.0f;
    if (tid == 0) smLds = 0.0f;

    // ---- stage A tile: contiguous 16*512 f32 chunk of x, b128 loads,
    //      packed bf16 conversion, b64 LDS stores ----
    {
        const float4* src4 = (const float4*)(x + (size_t)r0 * Cc);
        #pragma unroll
        for (int j = 0; j < 8; ++j) {
            int q = tid + 256 * j;              // 0..2047 float4's
            float4 v = src4[q];
            int r = q >> 7, c4 = q & 127;       // 128 float4 per 512-float row
            uint2 u;
            u.x = pack_bf16(v.x, v.y);
            u.y = pack_bf16(v.z, v.w);
            *(uint2*)&As[r * 520 + c4 * 4] = u; // 8B aligned (1040, 8 | both)
        }
    }

    const int m  = lane & 15;                   // M row within tile (A frag)
    const int kb = (lane >> 4) * 8;             // per-lane K sub-base (0 or 8)
    const int col0 = wv * 32 + (lane & 15);     // global H column, N-tile 0
    const int col1 = col0 + 16;                 // N-tile 1

    v8f acc0 = {}; v8f acc1 = {};
    const size_t headBase = (size_t)head * Cc * Hc;

    for (int kk = 0; kk < Cc; kk += 32) {
        // ---- stage B tile transposed: Bs[col][kr]. Each thread: 2 K-rows x
        //      4 cols per step -> packed (k,k+1) dwords, b128 global loads ----
        {
            const float* srcBase = w1 + headBase + (size_t)kk * Hc;
            #pragma unroll
            for (int j = 0; j < 4; ++j) {
                int p   = tid + 256 * j;        // 0..1023
                int c4  = p & 63;               // cols c4*4 .. c4*4+3
                int kr2 = p >> 6;               // rows 2*kr2, 2*kr2+1
                const float4 va = *(const float4*)(srcBase + (size_t)(2 * kr2)     * Hc + c4 * 4);
                const float4 vb = *(const float4*)(srcBase + (size_t)(2 * kr2 + 1) * Hc + c4 * 4);
                int colb = c4 * 4, krr = 2 * kr2;
                *(unsigned*)&Bs[(colb + 0) * 40 + krr] = pack_bf16(va.x, vb.x);
                *(unsigned*)&Bs[(colb + 1) * 40 + krr] = pack_bf16(va.y, vb.y);
                *(unsigned*)&Bs[(colb + 2) * 40 + krr] = pack_bf16(va.z, vb.z);
                *(unsigned*)&Bs[(colb + 3) * 40 + krr] = pack_bf16(va.w, vb.w);
            }
            if (kk + 32 < Cc)
                __builtin_prefetch(w1 + headBase + (size_t)(kk + 32) * Hc + tid, 0, 0);
        }
        __syncthreads();

        // ---- build fragments (two ds_load_b128 each) ----
        v8bf alo  = *(const v8bf*)&As[m * 520 + kk + kb];
        v8bf ahi  = *(const v8bf*)&As[m * 520 + kk + kb + 16];
        v8bf b0lo = *(const v8bf*)&Bs[col0 * 40 + kb];
        v8bf b0hi = *(const v8bf*)&Bs[col0 * 40 + kb + 16];
        v8bf b1lo = *(const v8bf*)&Bs[col1 * 40 + kb];
        v8bf b1hi = *(const v8bf*)&Bs[col1 * 40 + kb + 16];

        v16bf af, bf0, bf1;
        #pragma unroll
        for (int e = 0; e < 8; ++e) {
            af[e]  = alo[e];  af[e + 8]  = ahi[e];
            bf0[e] = b0lo[e]; bf0[e + 8] = b0hi[e];
            bf1[e] = b1lo[e]; bf1[e + 8] = b1hi[e];
        }

        acc0 = __builtin_amdgcn_wmma_f32_16x16x32_bf16(false, af, false, bf0,
                                                       (short)0, acc0, false, false);
        acc1 = __builtin_amdgcn_wmma_f32_16x16x32_bf16(false, af, false, bf1,
                                                       (short)0, acc1, false, false);
        __syncthreads();
    }

    // ---- epilogue: bias + gelu + fused h@w2 and h.ws reductions ----
    const float b1a = b1[head * Hc + col0], b1b = b1[head * Hc + col1];
    const float w20a = w2[col0 * 2 + 0], w21a = w2[col0 * 2 + 1];
    const float w20b = w2[col1 * 2 + 0], w21b = w2[col1 * 2 + 1];
    const float wsa = wsv[col0], wsb = wsv[col1];

    float pm[8], ps[8], sm = 0.0f;
    #pragma unroll
    for (int j = 0; j < 8; ++j) {
        float g0 = gelu_exact(acc0[j] + b1a);
        float g1 = gelu_exact(acc1[j] + b1b);
        pm[j] = g0 * w20a + g1 * w20b;
        ps[j] = g0 * w21a + g1 * w21b;
        sm   += g0 * wsa  + g1 * wsb;
    }

    // reduce over the 16 columns (lanes 0-15 / 16-31 hold the same rows)
    #pragma unroll
    for (int msk = 1; msk < 16; msk <<= 1) {
        #pragma unroll
        for (int j = 0; j < 8; ++j) {
            pm[j] += __shfl_xor(pm[j], msk, 32);
            ps[j] += __shfl_xor(ps[j], msk, 32);
        }
    }
    #pragma unroll
    for (int msk = 1; msk < 32; msk <<= 1) sm += __shfl_xor(sm, msk, 32);

    if ((lane & 15) == 0) {
        int rbase = (lane >> 4) * 8;            // lane0 -> rows 0..7, lane16 -> 8..15
        #pragma unroll
        for (int j = 0; j < 8; ++j) {
            atomicAdd(&muRow[(rbase + j) * 2 + 0], pm[j]);
            atomicAdd(&muRow[(rbase + j) * 2 + 1], ps[j]);
        }
    }
    if (lane == 0) atomicAdd(&smLds, sm);
    __syncthreads();

    if (tid < 32) {
        int row = tid >> 1, e = tid & 1;
        float v = muRow[tid] + b2[e];
        muSig[((size_t)(b * Tc + t0 + row) * NOc + head) * 2 + e] = v;
    }
    if (tid == 0) atomicAdd(&smacc[b * NOc + head], smLds);
}

// ---------------------------------------------------------------------------
// Phase 2: fused noise generation + per-sample Gaussian smoothing along T.
// One thread owns one spatial column s, slides a 17-tap window over t.
// Writes are coalesced across s (consecutive lanes -> consecutive s).
// ---------------------------------------------------------------------------
static __device__ __forceinline__ unsigned pcg32(unsigned v) {
    v = v * 747796405u + 2891336453u;
    unsigned w = ((v >> ((v >> 28) + 4u)) ^ v) * 277803737u;
    return (w >> 22) ^ w;
}

static __device__ __forceinline__ float gaussian_eps(unsigned key) {
    unsigned a = pcg32(key);
    unsigned c = pcg32(key ^ 0x9E3779B9u);
    float u1 = ((a >> 8) + 1u) * 5.96046448e-08f;   // (0,1]
    float u2 = (c >> 8) * 5.96046448e-08f;          // [0,1)
    return sqrtf(-2.0f * __logf(u1)) * __cosf(6.28318531f * u2);
}

static __device__ __forceinline__ int refl(int t) {
    t = (t < 0) ? -t : t;
    return (t >= Tc) ? (2 * Tc - 2 - t) : t;
}

// msBase = muSig + (b*Tc*NOc + head)*2 ; element for time tt at msBase[tt*NOc*2]
static __device__ __forceinline__ float sample_n(const float* __restrict__ msBase,
                                                 int b, int tt, int head, int S2, int s) {
    float mu = msBase[tt * (NOc * 2)];
    float si = msBase[tt * (NOc * 2) + 1];
    unsigned key = ((unsigned)(b * Tc + tt) * (unsigned)S2 + (unsigned)s) * 0x9E3779B1u
                 ^ (unsigned)(head * 0x85EBCA6B + 0x1234567);
    return mu + si * gaussian_eps(key);
}

__global__ void __launch_bounds__(256)
noise_smooth_kernel(const float* __restrict__ muSig,
                    const float* __restrict__ smacc,
                    const float* __restrict__ bsp,
                    float* __restrict__ outBase,
                    int head, int S2, int G, int SC)
{
    constexpr int TCH = 128, NT = Tc / TCH;
    int bx = blockIdx.x;
    int tch = bx % NT;  bx /= NT;
    int sch = bx % SC;  bx /= SC;
    int bgrp = bx;
    int tid = threadIdx.x;

    int b, s;
    if (S2 >= 256) { b = bgrp;               s = sch * 256 + tid; }
    else           { b = bgrp * G + tid / S2; s = tid % S2; }

    // per-sample smoothing kernel
    float sg = smacc[b * NOc + head] * (1.0f / (float)Tc) + bsp[0];
    float denom = sg * sg + 1e-6f;
    float kn[17]; float ksum = 0.0f;
    #pragma unroll
    for (int j = 0; j < 17; ++j) {
        float d = (float)(j - Rr);
        float kv = __expf(-0.5f * d * d / denom);
        kn[j] = kv; ksum += kv;
    }
    float inv = 1.0f / ksum;
    #pragma unroll
    for (int j = 0; j < 17; ++j) kn[j] *= inv;

    const float* msBase = muSig + ((size_t)b * Tc * NOc + head) * 2;

    int t0 = tch * TCH;
    float w[17];
    #pragma unroll
    for (int p = 0; p < 17; ++p)
        w[p] = sample_n(msBase, b, refl(t0 - Rr + p), head, S2, s);

    float* outCol = outBase + (size_t)b * Tc * S2 + s;
    #pragma unroll 8
    for (int dt = 0; dt < TCH; ++dt) {
        int t = t0 + dt;
        float o = 0.0f;
        #pragma unroll
        for (int p = 0; p < 17; ++p) o += kn[p] * w[p];
        outCol[(size_t)t * S2] = o;
        #pragma unroll
        for (int p = 0; p < 16; ++p) w[p] = w[p + 1];   // renamed away when unrolled
        w[16] = sample_n(msBase, b, refl(t + Rr + 1), head, S2, s);
    }
}

// ---------------------------------------------------------------------------
// Launcher
// ---------------------------------------------------------------------------
extern "C" void kernel_launch(void* const* d_in, const int* in_sizes, int n_in,
                              void* d_out, int out_size, void* d_ws, size_t ws_size,
                              hipStream_t stream) {
    const float* x   = (const float*)d_in[0];
    const float* w1  = (const float*)d_in[1];
    const float* b1  = (const float*)d_in[2];
    const float* w2  = (const float*)d_in[3];
    const float* b2  = (const float*)d_in[4];
    const float* wsv = (const float*)d_in[5];
    const float* bsp = (const float*)d_in[6];
    float* out = (float*)d_out;

    float* smacc = (float*)d_ws;           // 80 used, 128 reserved
    float* muSig = smacc + 128;            // B*T*NO*2 = 327680 floats

    noise_init_kernel<<<1, 128, 0, stream>>>(smacc);

    // Phase 1: 2048 M-tiles x 5 heads
    noise_gemm_wmma_kernel<<<(Tc * Bc / 16) * NOc, 256, 0, stream>>>(
        x, w1, b1, w2, b2, wsv, muSig, smacc);

    // Phase 2: one launch per head
    const int S2s[NOc]    = {16, 64, 256, 1024, 4096};
    const int outOff[NOc] = {0, 16, 80, 336, 1360};   // prefix sums, in units of B*T
    for (int i = 0; i < NOc; ++i) {
        int S2 = S2s[i];
        int G  = (S2 >= 256) ? 1 : (256 / S2);        // b's per block
        int SC = (S2 <= 256) ? 1 : (S2 / 256);        // s-chunks per b
        int grid = (Bc / G) * SC * (Tc / 128);
        noise_smooth_kernel<<<grid, 256, 0, stream>>>(
            muSig, smacc, bsp, out + (size_t)outOff[i] * Bc * Tc,
            i, S2, G, SC);
    }
}
